// Gene_14448269983949
// MI455X (gfx1250) — matrix operand order repaired and checked by reading, not compiled, not run on hardware
//
#include <hip/hip_runtime.h>

// ---------------------------------------------------------------------------
// CopyNet decoder step for MI455X (gfx1250, wave32).
//  - Heavy matmuls (enc_output @ W, 2 x 68.7 GFLOP) -> v_wmma_f32_16x16x32_bf16
//    with weights PREPACKED once into bf16 WMMA B-fragment order (one 32B load
//    per fragment) and enc tiles staged in LDS in A-fragment order.
//    M-tile = 64 rows/WG so each B fragment feeds 4 WMMAs and total B L2
//    traffic is 1.07 GB (vs 4.3 GB at M=16) -- below the 256MB-stream HBM floor.
//  - Logits matmul (state @ dense_W, softmax-sensitive) -> v_wmma_f32_16x16x4_f32
//    full f32 precision, dense_W (102 MB) streamed exactly once.
//  - Everything else memory-bound VALU with branch-free fast exp/tanh.
// HBM floor ~875 MB => ~38 us at 23.3 TB/s.
// ---------------------------------------------------------------------------

#define B_    64
#define S_    2048
#define U_    512
#define EMB_  256
#define V_    50000
#define OOV_  64
#define DEPTH_ (V_ + OOV_)       // 50064
#define G3U   (3 * U_)           // 1536
#define NEG_BIG -1e10f

typedef __attribute__((ext_vector_type(16))) __bf16          v16bf;
typedef __attribute__((ext_vector_type(8)))  float           v8f;
typedef __attribute__((ext_vector_type(2)))  float           v2f;

// pack two f32 -> packed bf16x2 (round-to-nearest-even), low = a
__device__ __forceinline__ unsigned int pack2bf(float a, float b) {
  unsigned int ua = __float_as_uint(a), ub = __float_as_uint(b);
  ua = ua + 0x7FFFu + ((ua >> 16) & 1u);
  ub = ub + 0x7FFFu + ((ub >> 16) & 1u);
  return (ua >> 16) | (ub & 0xFFFF0000u);
}

// branch-free tanh: saturates correctly (exp->inf => 1, exp->0 => -1), no NaN
__device__ __forceinline__ float tanh_fast(float x) {
  float e = __expf(2.f * x);
  return 1.f - 2.f / (e + 1.f);
}
__device__ __forceinline__ float sigmoid_fast(float x) {
  return 1.f / (1.f + __expf(-x));
}

// ---------------------------------------------------------------------------
__global__ void k_zero(float* __restrict__ p, int n) {
  int i = blockIdx.x * 256 + threadIdx.x;
  if (i < n) p[i] = 0.f;
}

__global__ void k_gather_emb(const int* __restrict__ x,
                             const float* __restrict__ w_emd,
                             float* __restrict__ dec_emb) {
  int i = blockIdx.x * 256 + threadIdx.x;        // B*EMB
  if (i >= B_ * EMB_) return;
  int b = i / EMB_, e = i % EMB_;
  int id = x[b]; if (id >= V_) id = V_ - 1;
  dec_emb[i] = w_emd[(size_t)id * EMB_ + e];
}

// out[b,j] = sum_k hidden[b,k] * W[k,j] + bias[j]   (K = 512 fixed)
__global__ void k_matvec(const float* __restrict__ h, const float* __restrict__ W,
                         const float* __restrict__ bias, float* __restrict__ out,
                         int N) {
  int i = blockIdx.x * 256 + threadIdx.x;
  if (i >= B_ * N) return;
  int b = i / N, j = i % N;
  const float* hb = h + b * U_;
  float acc = bias[j];
  #pragma unroll 4
  for (int k = 0; k < U_; ++k) acc += hb[k] * W[(size_t)k * N + j];
  out[i] = acc;
}

// selective_read[b,n] = sum_s enc_output[b,s,n] * (enc_inputs[b,s]==x[b])
__global__ void k_selread(const int* __restrict__ enc_inputs,
                          const int* __restrict__ x,
                          const float* __restrict__ eo,
                          float* __restrict__ selread) {
  int n = blockIdx.x * 256 + threadIdx.x;        // grid.x = 2
  int b = blockIdx.y;
  int xv = x[b];
  const int*   ei = enc_inputs + b * S_;
  const float* eb = eo + (size_t)b * S_ * U_;
  float acc = 0.f;
  for (int s = 0; s < S_; ++s)
    if (ei[s] == xv) acc += eb[(size_t)s * U_ + n];
  selread[b * U_ + n] = acc;
}

// ---------------------------------------------------------------------------
// Prepack a 512x512 f32 weight (row-major, [k,n]) into bf16 WMMA B-fragment
// order: pre_dword[((nt*16 + kk)*32 + lane)*8 + pair] = pack(W[k0,n],W[k0+1,n])
// where n = nt*16 + (lane&15), k0 = kk*32 + (lane>=16?16:0) + 2*pair.
// Each fragment is then one contiguous 32B per lane in the GEMM hot loop.
// ---------------------------------------------------------------------------
__global__ void k_prepack_w(const float* __restrict__ W, unsigned int* __restrict__ pre) {
  int i = blockIdx.x * 256 + threadIdx.x;        // 32*16*32*8 = 131072
  int lane = i & 31;
  int pair = (i >> 5) & 7;
  int kk   = (i >> 8) & 15;
  int nt   = i >> 12;                            // 0..31
  int n  = nt * 16 + (lane & 15);
  int k0 = kk * 32 + ((lane >> 4) << 4) + 2 * pair;
  unsigned int d = pack2bf(W[(size_t)k0 * U_ + n], W[(size_t)(k0 + 1) * U_ + n]);
  pre[((nt * 16 + kk) * 32 + lane) * 8 + pair] = d;
}

// ---------------------------------------------------------------------------
// Fused: out_score[b,s] = maybe_mask( tanh(eo[b,s,:] @ W + bias + rowadd[b,:])
//                                     . dvec + dbias )
// One 512-thread WG per (b, 64 s-rows). 16 waves; wave w owns n-tiles
// {2w, 2w+1} x m-subtiles {0..3}: each B fragment feeds 4 WMMAs.
// A tile staged in LDS in WMMA A-fragment order (32B/lane contiguous);
// the LDS block is reused for the epilogue reduction after a barrier.
// ---------------------------------------------------------------------------
__global__ void __launch_bounds__(512)
k_tanh_dot(const float* __restrict__ eo, const unsigned short* __restrict__ preW,
           const float* __restrict__ bias, const float* __restrict__ rowadd,
           const float* __restrict__ dvec, int dvec_per_b,
           const float* __restrict__ dbias,    // 1 element or nullptr
           const float* __restrict__ seqlen,   // [B] or nullptr
           float* __restrict__ out_score) {
  __shared__ unsigned int sMem[17408];           // 68 KB: A frags, then reduction
  unsigned int* sAi = sMem;                      // 4*16*32*8 = 16384 dwords
  float* sP = (float*)sMem;                      // 64 rows x 272 stride

  int tid = threadIdx.x;
  int b   = blockIdx.y;
  int s0  = blockIdx.x * 64;

  // ---- stage 64x512 f32 -> bf16 LDS in A-fragment order --------------------
  const float2* eb2 = (const float2*)(eo + (size_t)b * S_ * U_ + (size_t)s0 * U_);
  {
    int kp = tid & 255;                          // k-pair index, loop-invariant
    int mb = tid >> 8;                           // 0/1
    int kk = kp >> 4;
    int kr = (2 * kp) & 31;
    int halfS = kr >> 4, h = kr & 15;
    #pragma unroll 4
    for (int it = 0; it < 32; ++it) {
      int m = it * 2 + mb;                       // 0..63
      float2 v = eb2[m * 256 + kp];              // coalesced
      int mt = m >> 4, mr = m & 15;
      int lane = halfS * 16 + mr;
      sAi[((mt * 16 + kk) * 32 + lane) * 8 + (h >> 1)] = pack2bf(v.x, v.y);
    }
  }
  __syncthreads();

  // ---- GEMM: per kk = 4 LDS 32B loads + 2 global 32B loads + 8 WMMA --------
  int wave = tid >> 5, lane = tid & 31;
  int halfL = lane >> 4, l16 = lane & 15;
  v8f acc[4][2] = {{{}, {}}, {{}, {}}, {{}, {}}, {{}, {}}};
  for (int kk = 0; kk < 16; ++kk) {
    v16bf A[4];
    #pragma unroll
    for (int mt = 0; mt < 4; ++mt)
      A[mt] = *(const v16bf*)(sAi + ((mt * 16 + kk) * 32 + lane) * 8);
    v16bf Bv[2];
    #pragma unroll
    for (int nt = 0; nt < 2; ++nt) {
      int ntg = wave * 2 + nt;
      Bv[nt] = *(const v16bf*)(preW + ((size_t)(ntg * 16 + kk) * 32 + lane) * 16);
    }
    #pragma unroll
    for (int mt = 0; mt < 4; ++mt)
      #pragma unroll
      for (int nt = 0; nt < 2; ++nt)
        acc[mt][nt] = __builtin_amdgcn_wmma_f32_16x16x32_bf16(
            false, A[mt], false, Bv[nt], (short)0, acc[mt][nt], false, false);
  }
  __syncthreads();                               // done reading sAi

  // ---- epilogue: tanh + dot with dvec, deterministic reduction -------------
  const float* dv = dvec_per_b ? (dvec + b * U_) : dvec;
  float bb[2], dvn[2];
  #pragma unroll
  for (int nt = 0; nt < 2; ++nt) {
    int n = wave * 32 + nt * 16 + l16;
    bb[nt] = bias[n];
    if (rowadd) bb[nt] += rowadd[b * U_ + n];
    dvn[nt] = dv[n];
  }
  #pragma unroll
  for (int mt = 0; mt < 4; ++mt) {
    #pragma unroll
    for (int r = 0; r < 8; ++r) {                // C/D layout: M = r + (lane<16?0:8)
      float part = 0.f;
      #pragma unroll
      for (int nt = 0; nt < 2; ++nt)
        part += tanh_fast(acc[mt][nt][r] + bb[nt]) * dvn[nt];
      int m = mt * 16 + r + halfL * 8;
      sP[m * 272 + wave * 16 + l16] = part;
    }
  }
  __syncthreads();
  if (tid < 64) {
    float sum = dbias ? dbias[0] : 0.f;
    for (int j = 0; j < 256; ++j) sum += sP[tid * 272 + j];
    if (seqlen && !((float)(s0 + tid) < seqlen[b])) sum = NEG_BIG;
    out_score[b * S_ + s0 + tid] = sum;
  }
}

// ---------------------------------------------------------------------------
// attention softmax over S per batch row + seq_len count
__global__ void k_attn_softmax(const float* __restrict__ score,
                               float* __restrict__ attn,
                               const int* __restrict__ enc_inputs,
                               float* __restrict__ seqlen) {
  __shared__ float red[256];
  int b = blockIdx.x, tid = threadIdx.x;
  const float* sc = score + b * S_;
  float mx = -3.4e38f;
  for (int s = tid; s < S_; s += 256) mx = fmaxf(mx, sc[s]);
  red[tid] = mx; __syncthreads();
  for (int o = 128; o > 0; o >>= 1) { if (tid < o) red[tid] = fmaxf(red[tid], red[tid + o]); __syncthreads(); }
  mx = red[0]; __syncthreads();
  float sm = 0.f;
  for (int s = tid; s < S_; s += 256) sm += __expf(sc[s] - mx);
  red[tid] = sm; __syncthreads();
  for (int o = 128; o > 0; o >>= 1) { if (tid < o) red[tid] += red[tid + o]; __syncthreads(); }
  sm = red[0]; __syncthreads();
  float inv = 1.f / sm;
  for (int s = tid; s < S_; s += 256) attn[b * S_ + s] = __expf(sc[s] - mx) * inv;
  int cnt = 0;
  for (int s = tid; s < S_; s += 256) cnt += (enc_inputs[b * S_ + s] > 0) ? 1 : 0;
  red[tid] = (float)cnt; __syncthreads();
  for (int o = 128; o > 0; o >>= 1) { if (tid < o) red[tid] += red[tid + o]; __syncthreads(); }
  if (tid == 0) seqlen[b] = red[0];
}

// context = sum_s attn[b,s] * eo[b,s,:]  -- deterministic partials, 16 s-chunks
#define CZ 16
__global__ void k_ctx_part(const float* __restrict__ attn, const float* __restrict__ eo,
                           float* __restrict__ part) {
  int n = blockIdx.x * 256 + threadIdx.x;        // grid.x = 2
  int b = blockIdx.y, z = blockIdx.z;
  int s0 = z * (S_ / CZ);
  const float* ab = attn + b * S_;
  const float* ebase = eo + (size_t)b * S_ * U_;
  float acc = 0.f;
  for (int s = s0; s < s0 + S_ / CZ; ++s)
    acc += ab[s] * ebase[(size_t)s * U_ + n];
  part[((size_t)z * B_ + b) * U_ + n] = acc;
}
__global__ void k_ctx_reduce(const float* __restrict__ part, float* __restrict__ ctx) {
  int i = blockIdx.x * 256 + threadIdx.x;
  if (i >= B_ * U_) return;
  float acc = 0.f;
  for (int z = 0; z < CZ; ++z) acc += part[(size_t)z * B_ * U_ + i];
  ctx[i] = acc;
}

// xg[b,j] = [context, selread, dec_emb] @ gru_kernel + gru_bias[0]
__global__ void k_gru_xg(const float* __restrict__ ctx, const float* __restrict__ selread,
                         const float* __restrict__ dec_emb, const float* __restrict__ gk,
                         const float* __restrict__ gbias0, float* __restrict__ xg) {
  int i = blockIdx.x * 256 + threadIdx.x;
  if (i >= B_ * G3U) return;
  int b = i / G3U, j = i % G3U;
  const float* c  = ctx + b * U_;
  const float* sr = selread + b * U_;
  const float* de = dec_emb + b * EMB_;
  float acc = gbias0[j];
  for (int k = 0; k < U_;   ++k) acc += c[k]  * gk[(size_t)k * G3U + j];
  for (int k = 0; k < U_;   ++k) acc += sr[k] * gk[(size_t)(U_ + k) * G3U + j];
  for (int k = 0; k < EMB_; ++k) acc += de[k] * gk[(size_t)(2 * U_ + k) * G3U + j];
  xg[i] = acc;
}

__global__ void k_gru_state(const float* __restrict__ xg, const float* __restrict__ hg,
                            const float* __restrict__ hidden,
                            float* __restrict__ state_ws, float* __restrict__ out_state) {
  int i = blockIdx.x * 256 + threadIdx.x;
  if (i >= B_ * U_) return;
  int b = i / U_, u = i % U_;
  const float* xb = xg + b * G3U;
  const float* hb = hg + b * G3U;
  float z  = sigmoid_fast(xb[u] + hb[u]);
  float r  = sigmoid_fast(xb[U_ + u] + hb[U_ + u]);
  float hh = tanh_fast(xb[2 * U_ + u] + r * hb[2 * U_ + u]);
  float st = z * hidden[i] + (1.f - z) * hh;
  state_ws[i] = st;
  out_state[i] = st;
}

// ---------------------------------------------------------------------------
// generate_score = state(64x512) @ dense_W(512x50000) + dense_b
// Full-precision f32 WMMA (16x16x4). One wave per 16-column tile; the wave
// computes all four 16-row b-tiles so dense_W (102 MB) is streamed once.
// ---------------------------------------------------------------------------
__global__ void __launch_bounds__(256)
k_gen_score(const float* __restrict__ state, const float* __restrict__ dW,
            const float* __restrict__ db, float* __restrict__ gs) {
  int wave = threadIdx.x >> 5, lane = threadIdx.x & 31;
  int vt = blockIdx.x * 8 + wave;
  if (vt >= V_ / 16) return;                      // 3125 tiles; uniform per wave
  int halfL = lane >> 4, l16 = lane & 15;
  int n = vt * 16 + l16;
  v8f acc[4] = {{}, {}, {}, {}};
  for (int k0 = 0; k0 < U_; k0 += 4) {
    int kb = k0 + halfL * 2;                      // f32: lanes0-15 K={0,1}, 16-31 K={2,3}
    v2f bf;
    bf.x = dW[(size_t)kb * V_ + n];
    bf.y = dW[(size_t)(kb + 1) * V_ + n];
    #pragma unroll
    for (int t = 0; t < 4; ++t) {
      int m = t * 16 + l16;
      v2f af;
      af.x = state[m * U_ + kb];
      af.y = state[m * U_ + kb + 1];
      acc[t] = __builtin_amdgcn_wmma_f32_16x16x4_f32(
          false, af, false, bf, (short)0, acc[t], false, false);
    }
  }
  float bias = db[n];
  #pragma unroll
  for (int t = 0; t < 4; ++t) {
    #pragma unroll
    for (int r = 0; r < 8; ++r) {
      int brow = t * 16 + r + halfL * 8;
      gs[(size_t)brow * V_ + n] = acc[t][r] + bias;
    }
  }
}

// row max & expsum over concat [gen_score(50000), copy_score(2048)]
__global__ void k_final_stats(const float* __restrict__ gs, const float* __restrict__ cs,
                              float* __restrict__ rmax, float* __restrict__ rsum) {
  __shared__ float red[256];
  int b = blockIdx.x, tid = threadIdx.x;
  const float* g = gs + (size_t)b * V_;
  const float* c = cs + b * S_;
  float mx = -3.4e38f;
  for (int i = tid; i < V_; i += 256) mx = fmaxf(mx, g[i]);
  for (int i = tid; i < S_; i += 256) mx = fmaxf(mx, c[i]);
  red[tid] = mx; __syncthreads();
  for (int o = 128; o > 0; o >>= 1) { if (tid < o) red[tid] = fmaxf(red[tid], red[tid + o]); __syncthreads(); }
  mx = red[0]; __syncthreads();
  float sm = 0.f;
  for (int i = tid; i < V_; i += 256) sm += __expf(g[i] - mx);
  for (int i = tid; i < S_; i += 256) sm += __expf(c[i] - mx);
  red[tid] = sm; __syncthreads();
  for (int o = 128; o > 0; o >>= 1) { if (tid < o) red[tid] += red[tid + o]; __syncthreads(); }
  if (tid == 0) { rmax[b] = mx; rsum[b] = red[0]; }
}

// scatter prob_c into out[b, enc_inputs[b,s]]
__global__ void k_scatter(const float* __restrict__ cs, const int* __restrict__ enc_inputs,
                          const float* __restrict__ rmax, const float* __restrict__ rsum,
                          float* __restrict__ out) {
  int i = blockIdx.x * 256 + threadIdx.x;
  if (i >= B_ * S_) return;
  int b = i / S_;
  float p = __expf(cs[i] - rmax[b]) / rsum[b];
  atomicAdd(&out[(size_t)b * DEPTH_ + enc_inputs[i]], p);
}

// out[b,v<V] += prob_g + 1e-10 ; out[b,v>=V] += 1e-10
__global__ void k_final_out(const float* __restrict__ gs, const float* __restrict__ rmax,
                            const float* __restrict__ rsum, float* __restrict__ out) {
  size_t i = (size_t)blockIdx.x * 256 + threadIdx.x;
  if (i >= (size_t)B_ * DEPTH_) return;
  int b = (int)(i / DEPTH_), v = (int)(i % DEPTH_);
  float add = 1e-10f;
  if (v < V_) add += __expf(gs[(size_t)b * V_ + v] - rmax[b]) / rsum[b];
  out[i] += add;
}

// ---------------------------------------------------------------------------
extern "C" void kernel_launch(void* const* d_in, const int* in_sizes, int n_in,
                              void* d_out, int out_size, void* d_ws, size_t ws_size,
                              hipStream_t stream) {
  const int*   x           = (const int*)  d_in[0];
  const int*   enc_inputs  = (const int*)  d_in[1];
  const float* eo          = (const float*)d_in[2];
  const float* hidden      = (const float*)d_in[3];
  const float* w_emd       = (const float*)d_in[4];
  const float* att_W1      = (const float*)d_in[5];
  const float* att_b1      = (const float*)d_in[6];
  const float* att_W2      = (const float*)d_in[7];
  const float* att_b2      = (const float*)d_in[8];
  const float* att_V       = (const float*)d_in[9];
  const float* att_bV      = (const float*)d_in[10];
  const float* gru_kernel  = (const float*)d_in[11];
  const float* gru_rec_k   = (const float*)d_in[12];
  const float* gru_bias    = (const float*)d_in[13];
  const float* dense_W     = (const float*)d_in[14];
  const float* dense_b     = (const float*)d_in[15];
  const float* dh_W        = (const float*)d_in[16];
  const float* dh_b        = (const float*)d_in[17];

  float* ws = (float*)d_ws;
  // workspace layout (floats)
  float* h2       = ws;                       // 64*512
  float* hg       = h2       + B_ * U_;       // 64*1536
  float* score    = hg       + B_ * G3U;      // 64*2048
  float* attn     = score    + B_ * S_;       // 64*2048
  float* selread  = attn     + B_ * S_;       // 64*512
  float* ctx      = selread  + B_ * U_;       // 64*512
  float* dec_emb  = ctx      + B_ * U_;       // 64*256
  float* xg       = dec_emb  + B_ * EMB_;     // 64*1536
  float* state_ws = xg       + B_ * G3U;      // 64*512
  float* copy_sc  = state_ws + B_ * U_;       // 64*2048
  float* rmax     = copy_sc  + B_ * S_;       // 64
  float* rsum     = rmax     + 128;           // 64 (padded)
  float* seqlen   = rsum     + 128;           // 64 (padded)
  float* ctx_part = seqlen   + 128;           // 16*64*512
  float* gen_sc   = ctx_part + (size_t)CZ * B_ * U_;       // 64*50000
  float* preW1f   = gen_sc   + (size_t)B_ * V_;            // 512*512 bf16 = 128K floats
  float* preWdhf  = preW1f   + (U_ * U_ / 2);              // 512*512 bf16
  unsigned int* preW1  = (unsigned int*)preW1f;
  unsigned int* preWdh = (unsigned int*)preWdhf;

  float* out_prob  = (float*)d_out;                        // 64 * 50064
  float* out_state = out_prob + (size_t)B_ * DEPTH_;       // 64 * 512

  // 0. one-time weight prepack into bf16 WMMA B-fragment order
  k_prepack_w<<<(U_ * U_ / 2 + 255) / 256, 256, 0, stream>>>(att_W1, preW1);
  k_prepack_w<<<(U_ * U_ / 2 + 255) / 256, 256, 0, stream>>>(dh_W, preWdh);

  // 1. small precompute
  k_gather_emb<<<(B_ * EMB_ + 255) / 256, 256, 0, stream>>>(x, w_emd, dec_emb);
  k_matvec<<<(B_ * U_  + 255) / 256, 256, 0, stream>>>(hidden, att_W2, att_b2, h2, U_);
  k_matvec<<<(B_ * G3U + 255) / 256, 256, 0, stream>>>(hidden, gru_rec_k, gru_bias + G3U, hg, G3U);

  // 2. attention scores: tanh(eo@W1 + b1 + h2) . att_V + bV   [bf16 WMMA]
  k_tanh_dot<<<dim3(S_ / 64, B_), 512, 0, stream>>>(
      eo, (const unsigned short*)preW1, att_b1, h2, att_V, 0, att_bV, nullptr, score);

  // 3. selective read (mask-gated scan)
  k_selread<<<dim3(U_ / 256, B_), 256, 0, stream>>>(enc_inputs, x, eo, selread);

  // 4. softmax over S + seq_len
  k_attn_softmax<<<B_, 256, 0, stream>>>(score, attn, enc_inputs, seqlen);

  // 5. context (deterministic chunked reduction)
  k_ctx_part<<<dim3(U_ / 256, B_, CZ), 256, 0, stream>>>(attn, eo, ctx_part);
  k_ctx_reduce<<<(B_ * U_ + 255) / 256, 256, 0, stream>>>(ctx_part, ctx);

  // 6. GRU
  k_gru_xg<<<(B_ * G3U + 255) / 256, 256, 0, stream>>>(ctx, selread, dec_emb,
                                                       gru_kernel, gru_bias, xg);
  k_gru_state<<<(B_ * U_ + 255) / 256, 256, 0, stream>>>(xg, hg, hidden, state_ws, out_state);

  // 7. generate scores: state @ dense_W + b   [f32 WMMA, exact]
  k_gen_score<<<(V_ / 16 + 7) / 8, 256, 0, stream>>>(state_ws, dense_W, dense_b, gen_sc);

  // 8. copy scores: tanh(eo@dh_W + dh_b) . state, masked   [bf16 WMMA]
  k_tanh_dot<<<dim3(S_ / 64, B_), 512, 0, stream>>>(
      eo, (const unsigned short*)preWdh, dh_b, nullptr, state_ws, 1, nullptr, seqlen, copy_sc);

  // 9. joint softmax over [gen(50000) | copy(2048)] + scatter + combine
  k_final_stats<<<B_, 256, 0, stream>>>(gen_sc, copy_sc, rmax, rsum);
  k_zero<<<((int)((size_t)B_ * DEPTH_) + 255) / 256, 256, 0, stream>>>(out_prob, B_ * DEPTH_);
  k_scatter<<<(B_ * S_ + 255) / 256, 256, 0, stream>>>(copy_sc, enc_inputs, rmax, rsum, out_prob);
  k_final_out<<<((int)((size_t)B_ * DEPTH_) + 255) / 256, 256, 0, stream>>>(gen_sc, rmax, rsum, out_prob);
}
